// HoloformerEncoderLayer_39771397161452
// MI455X (gfx1250) — matrix operand order repaired and checked by reading, not compiled, not run on hardware
//
#include <hip/hip_runtime.h>

#define DIMS 1024
#define FFD  4096
#define SEQ  4096
#define BATCH 4
#define MTOK (BATCH * SEQ)          // 16384 tokens
#define NFREQ 513                    // rfft bins for n=1024
#define NFPAD 1028                   // floats per padded spectrum row (514 complex)
#define QKSTR 2048                   // fused q|k row stride
#define EPSLN 1e-5f
#define PI_F 3.14159265358979323846f

typedef __attribute__((ext_vector_type(16))) __bf16 v16bf;
typedef __attribute__((ext_vector_type(8)))  __bf16 v8bf;
typedef __attribute__((ext_vector_type(8)))  float  v8f;

// ---------------- helpers ----------------
__device__ __forceinline__ unsigned short f2bf(float f) {
  unsigned int u = __builtin_bit_cast(unsigned int, f);
  u += 0x7fffu + ((u >> 16) & 1u);          // round-to-nearest-even
  return (unsigned short)(u >> 16);
}

// async global->LDS copy, 16B per lane (ASYNCcnt-tracked, no VGPR round trip)
#define ASYNC_B128(ldsoff, goff, sbase)                                   \
  asm volatile("global_load_async_to_lds_b128 %0, %1, %2"                 \
               :: "v"(ldsoff), "v"(goff), "s"(sbase) : "memory")
#define WAIT_ASYNC_5() asm volatile("s_wait_asynccnt 0x5" ::: "memory")
#define WAIT_ASYNC_0() asm volatile("s_wait_asynccnt 0x0" ::: "memory")

// A fragment: 16-bit A 16x32 layout (ISA 7.12.2): lane L holds row M=L%16,
// halves [0..7] = K = (L/16)*8 + 0..7 ; halves [8..15] = 16 + (L/16)*8 + 0..7
__device__ __forceinline__ v16bf load_fragA(const unsigned short* p, int lh) {
  const v8bf lo = *(const v8bf*)(p + lh * 8);
  const v8bf hi = *(const v8bf*)(p + 16 + lh * 8);
  v16bf f;
#pragma unroll
  for (int i = 0; i < 8; ++i) { f[i] = lo[i]; f[i + 8] = hi[i]; }
  return f;
}

// B fragment: 32x16 (KxN): lane L holds col N=L%16,
// halves [0..7] = K = (L/16)*8 + 0..7 ; halves [8..15] = 16 + (L/16)*8 + 0..7
__device__ __forceinline__ v16bf load_fragB(const unsigned short* p, int lh) {
  const v8bf lo = *(const v8bf*)(p + lh * 8);
  const v8bf hi = *(const v8bf*)(p + 16 + lh * 8);
  v16bf f;
#pragma unroll
  for (int i = 0; i < 8; ++i) { f[i] = lo[i]; f[i + 8] = hi[i]; }
  return f;
}

// ---------------- f32 -> bf16 conversion (one-time staging) -----------------
__global__ __launch_bounds__(256) void cvt_bf16(
    const float* __restrict__ in, unsigned short* __restrict__ out, int n)
{
  int i = (blockIdx.x * 256 + threadIdx.x) * 8;
  if (i >= n) return;
  float4 a = *(const float4*)(in + i);
  float4 b = *(const float4*)(in + i + 4);
  uint4 v;
  v.x = (unsigned)f2bf(a.x) | ((unsigned)f2bf(a.y) << 16);
  v.y = (unsigned)f2bf(a.z) | ((unsigned)f2bf(a.w) << 16);
  v.z = (unsigned)f2bf(b.x) | ((unsigned)f2bf(b.y) << 16);
  v.w = (unsigned)f2bf(b.z) | ((unsigned)f2bf(b.w) << 16);
  *(uint4*)(out + i) = v;
}

__global__ void copy_f32(const float* __restrict__ in, float* __restrict__ out, int n) {
  int i = blockIdx.x * 256 + threadIdx.x;
  if (i < n) out[i] = in[i];
}

// ---------------- GEMM: C[M,N] = A[M,K] * B[N,K]^T + bias (opt relu) --------
// A,B bf16.  Double-buffered async global->LDS tile copies.
// Block tile 64x256 (8 waves = 2x4, wave tile 32x64), K-step 32.
#define GTM 64
#define GTN 256
#define GTK 32
#define LROW 40                       // ushort row stride (80B, 16B-aligned)
#define A_BYTES (GTM * LROW * 2)      // 5120
#define B_BYTES (GTN * LROW * 2)      // 20480
#define BUF_BYTES (A_BYTES + B_BYTES) // 25600

__device__ __forceinline__ v8f wmma_bf16(v16bf a, v16bf b, v8f c) {
  return __builtin_amdgcn_wmma_f32_16x16x32_bf16(false, a, false, b, (short)0, c, false, false);
}

__device__ __forceinline__ void gemm_tile_issue(
    unsigned base, const unsigned short* A, const unsigned short* B,
    unsigned ldsA, unsigned& goA, const unsigned* ldsB, unsigned* goB)
{
  ASYNC_B128(base + ldsA, goA, A);
#pragma unroll
  for (int t = 0; t < 4; ++t) ASYNC_B128(base + ldsB[t], goB[t], B);
  goA += GTK * 2;
#pragma unroll
  for (int t = 0; t < 4; ++t) goB[t] += GTK * 2;
}

__device__ __forceinline__ void gemm_tile_compute(
    const char* smem, int cur, int wm, int wn, int lm, int lh, v8f acc[2][4])
{
  const unsigned short* sA = (const unsigned short*)(smem + cur * BUF_BYTES);
  const unsigned short* sB = sA + A_BYTES / 2;
  const unsigned short* pa = sA + (wm + lm) * LROW;
  const unsigned short* pb = sB + (wn + lm) * LROW;
  v16bf a0 = load_fragA(pa, lh);
  v16bf a1 = load_fragA(pa + 16 * LROW, lh);
  v16bf b[4];
#pragma unroll
  for (int j = 0; j < 4; ++j) b[j] = load_fragB(pb + j * 16 * LROW, lh);
#pragma unroll
  for (int j = 0; j < 4; ++j) {
    acc[0][j] = wmma_bf16(a0, b[j], acc[0][j]);
    acc[1][j] = wmma_bf16(a1, b[j], acc[1][j]);
  }
}

__global__ __launch_bounds__(256) void gemm_bf16(
    const unsigned short* __restrict__ A, const unsigned short* __restrict__ B,
    const float* __restrict__ bias, void* __restrict__ Cout,
    int M, int N, int K, int relu, int cvt_out)
{
  extern __shared__ char smem[];
  const int tid = threadIdx.x;
  const int m0 = blockIdx.x * GTM;
  const int n0 = blockIdx.y * GTN;
  const int wave = tid >> 5;
  const int lane = tid & 31;
  const int wm = (wave & 1) * 32;   // 2 waves along M
  const int wn = (wave >> 1) * 64;  // 4 waves along N, 64 cols each
  const int lm = lane & 15;
  const int lh = lane >> 4;
  const unsigned lds0 = __builtin_amdgcn_groupstaticsize();   // dynamic-LDS base

  // per-thread async copy slots: A tile 64x32 bf16 = 256 x 16B (1/thread),
  //                              B tile 256x32 bf16 = 1024 x 16B (4/thread)
  unsigned ldsA, goA, ldsB[4], goB[4];
  {
    int r = tid >> 2, ch = tid & 3;
    ldsA = (unsigned)(r * (LROW * 2) + ch * 16);
    goA  = (unsigned)(((size_t)(m0 + r) * K + ch * 8) * 2);
  }
#pragma unroll
  for (int t = 0; t < 4; ++t) {
    int i = tid + t * 256, r = i >> 2, ch = i & 3;
    ldsB[t] = (unsigned)(A_BYTES + r * (LROW * 2) + ch * 16);
    goB[t]  = (unsigned)(((size_t)(n0 + r) * K + ch * 8) * 2);
  }

  v8f acc[2][4];
#pragma unroll
  for (int i = 0; i < 2; ++i)
#pragma unroll
    for (int j = 0; j < 4; ++j) acc[i][j] = (v8f){0,0,0,0,0,0,0,0};

  const int ntiles = K / GTK;       // >= 2 always here
  gemm_tile_issue(lds0, A, B, ldsA, goA, ldsB, goB);           // tile 0 -> buf0
  for (int kt = 0; kt < ntiles - 1; ++kt) {
    const int cur = kt & 1;
    gemm_tile_issue(lds0 + (1 - cur) * BUF_BYTES, A, B, ldsA, goA, ldsB, goB);
    WAIT_ASYNC_5();                 // 5 outstanding -> current tile complete
    __syncthreads();
    gemm_tile_compute(smem, cur, wm, wn, lm, lh, acc);
    __syncthreads();                // reads done before buffer reuse
  }
  WAIT_ASYNC_0();
  __syncthreads();
  gemm_tile_compute(smem, (ntiles - 1) & 1, wm, wn, lm, lh, acc);

#pragma unroll
  for (int i = 0; i < 2; ++i) {
#pragma unroll
    for (int j = 0; j < 4; ++j) {
      int mb = m0 + wm + 16 * i + 8 * lh;   // C layout: VGPR r -> M = r + 8*(lane/16)
      int nb = n0 + wn + 16 * j + lm;       //           N = lane%16
      float bv = bias[nb];
#pragma unroll
      for (int r = 0; r < 8; ++r) {
        float v = acc[i][j][r] + bv;
        if (relu) v = fmaxf(v, 0.0f);
        if (cvt_out) ((unsigned short*)Cout)[(size_t)(mb + r) * N + nb] = f2bf(v);
        else         ((float*)Cout)[(size_t)(mb + r) * N + nb] = v;
      }
    }
  }
}

// ---------------- FFT machinery (512-pt complex, Stockham radix-2) ----------
__device__ __forceinline__ float2 cmulf(float2 a, float2 b) {
  return make_float2(a.x * b.x - a.y * b.y, a.x * b.y + a.y * b.x);
}

__device__ float2* fft512(float2* a, float2* b, float sign) {
  const int t = threadIdx.x;                  // 256 threads == n/2 butterflies
  for (int p = 1; p < 512; p <<= 1) {
    __syncthreads();
    int k = t & (p - 1);
    float2 u = a[t];
    float2 v = a[t + 256];
    float ang = sign * PI_F * (float)k / (float)p;
    float sn, cs; __sincosf(ang, &sn, &cs);
    float2 vv = cmulf(make_float2(cs, sn), v);
    int j = (t << 1) - k;
    b[j]     = make_float2(u.x + vv.x, u.y + vv.y);
    b[j + p] = make_float2(u.x - vv.x, u.y - vv.y);
    float2* tmp = a; a = b; b = tmp;
  }
  __syncthreads();
  return a;
}

__device__ __forceinline__ float2 rfft_untangle(const float2* Z, int k) {
  int k1 = k & 511, k2 = (512 - k) & 511;
  float2 a = Z[k1], c = Z[k2];
  float2 E = make_float2(0.5f * (a.x + c.x), 0.5f * (a.y - c.y));
  float2 O = make_float2(0.5f * (a.y + c.y), 0.5f * (c.x - a.x));
  float ang = -PI_F * (float)k * (1.0f / 512.0f);
  float sn, cs; __sincosf(ang, &sn, &cs);
  return make_float2(E.x + cs * O.x - sn * O.y, E.y + cs * O.y + sn * O.x);
}

__device__ float2 block_reduce2(float2 v, float2* red) {
  const int tid = threadIdx.x;
  red[tid] = v;
  __syncthreads();
  for (int s = 128; s > 0; s >>= 1) {
    if (tid < s) { red[tid].x += red[tid + s].x; red[tid].y += red[tid + s].y; }
    __syncthreads();
  }
  float2 r = red[0];
  __syncthreads();
  return r;
}

// ---------------- Stage A: LN(q), LN(k), spectra, XB = X .* K ---------------
__global__ __launch_bounds__(256) void holo_stage_a(
    const float* __restrict__ x, const float* __restrict__ qk,
    const float* __restrict__ gq, const float* __restrict__ bq,
    const float* __restrict__ gk, const float* __restrict__ bk,
    float* __restrict__ XB, float* __restrict__ Qf)
{
  __shared__ __attribute__((aligned(16))) float2 fa[512];
  __shared__ __attribute__((aligned(16))) float2 fb[512];
  __shared__ float2 spX[NFREQ];
  __shared__ float row[DIMS];
  __shared__ float2 red[256];
  const int tid = threadIdx.x;
  const size_t tok = blockIdx.x;

  const float2* xr = (const float2*)(x + tok * DIMS);
  fa[tid] = xr[tid];
  fa[tid + 256] = xr[tid + 256];
  float2* Z = fft512(fa, fb, -1.0f);
  for (int k = tid; k <= 512; k += 256) spX[k] = rfft_untangle(Z, k);
  __syncthreads();

  {  // k branch
    const float* src = qk + tok * QKSTR + DIMS;
    float s = 0.f, ss = 0.f;
#pragma unroll
    for (int i = 0; i < 4; ++i) {
      int d = tid + (i << 8);
      float v = src[d];
      row[d] = v; s += v; ss += v * v;
    }
    float2 tot = block_reduce2(make_float2(s, ss), red);
    float mean = tot.x * (1.0f / DIMS);
    float inv  = rsqrtf(tot.y * (1.0f / DIMS) - mean * mean + EPSLN);
#pragma unroll
    for (int i = 0; i < 2; ++i) {
      int n = tid + (i << 8);
      int d = n << 1;
      float e0 = (row[d]     - mean) * inv * gk[d]     + bk[d];
      float e1 = (row[d + 1] - mean) * inv * gk[d + 1] + bk[d + 1];
      fa[n] = make_float2(e0, e1);
    }
    float2* Zk = fft512(fa, fb, -1.0f);
    float2* outp = (float2*)(XB + tok * NFPAD);
    for (int k = tid; k <= 512; k += 256) {
      float2 Kk = rfft_untangle(Zk, k);
      outp[k] = cmulf(spX[k], Kk);
    }
    __syncthreads();
  }

  {  // q branch
    const float* src = qk + tok * QKSTR;
    float s = 0.f, ss = 0.f;
#pragma unroll
    for (int i = 0; i < 4; ++i) {
      int d = tid + (i << 8);
      float v = src[d];
      row[d] = v; s += v; ss += v * v;
    }
    float2 tot = block_reduce2(make_float2(s, ss), red);
    float mean = tot.x * (1.0f / DIMS);
    float inv  = rsqrtf(tot.y * (1.0f / DIMS) - mean * mean + EPSLN);
#pragma unroll
    for (int i = 0; i < 2; ++i) {
      int n = tid + (i << 8);
      int d = n << 1;
      float e0 = (row[d]     - mean) * inv * gq[d]     + bq[d];
      float e1 = (row[d + 1] - mean) * inv * gq[d + 1] + bq[d + 1];
      fa[n] = make_float2(e0, e1);
    }
    float2* Zq = fft512(fa, fb, -1.0f);
    float2* outp = (float2*)(Qf + tok * NFPAD);
    for (int k = tid; k <= 512; k += 256) outp[k] = rfft_untangle(Zq, k);
  }
}

// ---------------- Frequency-domain sequence sum: S[b] = sum_s XB[b,s] -------
__global__ void holo_sum_xb(const float* __restrict__ XB, float* __restrict__ Sf) {
  int f = blockIdx.x * blockDim.x + threadIdx.x;
  int b = blockIdx.y;
  if (f < 2 * NFREQ) {
    const float* base = XB + (size_t)b * SEQ * NFPAD + f;
    float acc = 0.f;
    for (int s = 0; s < SEQ; ++s) acc += base[(size_t)s * NFPAD];
    Sf[(size_t)b * NFPAD + f] = acc;
  }
}

// ---------------- Stage B: mix = irfft(XB + S .* conj(Q)); LN(x + mix) ------
__global__ __launch_bounds__(256) void holo_stage_b(
    const float* __restrict__ x,
    const float* __restrict__ XB, const float* __restrict__ Qf,
    const float* __restrict__ Sf,
    const float* __restrict__ g0, const float* __restrict__ b0,
    float* __restrict__ out, unsigned short* __restrict__ x1bf)
{
  __shared__ __attribute__((aligned(16))) float2 fa[512];
  __shared__ __attribute__((aligned(16))) float2 fb[512];
  __shared__ float2 spF[NFREQ];
  __shared__ float row[DIMS];
  __shared__ float2 red[256];
  const int tid = threadIdx.x;
  const size_t tok = blockIdx.x;
  const int b = (int)(tok >> 12);   // tok / SEQ

  const float2* xbr = (const float2*)(XB + tok * NFPAD);
  const float2* qr  = (const float2*)(Qf + tok * NFPAD);
  const float2* sr  = (const float2*)(Sf + (size_t)b * NFPAD);
  for (int k = tid; k <= 512; k += 256) {
    float2 xb = xbr[k], q = qr[k], s = sr[k];
    float2 v = make_float2(s.x * q.x + s.y * q.y, s.y * q.x - s.x * q.y); // s*conj(q)
    spF[k] = make_float2(xb.x + v.x, xb.y + v.y);
  }
  __syncthreads();
  for (int k = tid; k < 512; k += 256) {
    float2 Fk = spF[k];
    float2 Fc = spF[512 - k];
    float2 E = make_float2(0.5f * (Fk.x + Fc.x), 0.5f * (Fk.y - Fc.y));
    float2 D = make_float2(0.5f * (Fk.x - Fc.x), 0.5f * (Fk.y + Fc.y));
    float ang = PI_F * (float)k * (1.0f / 512.0f);
    float sn, cs; __sincosf(ang, &sn, &cs);
    float2 O = cmulf(make_float2(cs, sn), D);
    fa[k] = make_float2(E.x - O.y, E.y + O.x);          // E + i*O
  }
  float2* Z = fft512(fa, fb, +1.0f);                    // inverse (unnormalized)

  const float* xrw = x + tok * DIMS;
  const float scale = 1.0f / 512.0f;
  float s = 0.f, ss = 0.f;
#pragma unroll
  for (int i = 0; i < 2; ++i) {
    int n = tid + (i << 8);
    float2 z = Z[n];
    int d = n << 1;
    float v0 = xrw[d]     + z.x * scale;
    float v1 = xrw[d + 1] + z.y * scale;
    row[d] = v0; row[d + 1] = v1;
    s += v0 + v1; ss += v0 * v0 + v1 * v1;
  }
  float2 tot = block_reduce2(make_float2(s, ss), red);
  float mean = tot.x * (1.0f / DIMS);
  float inv  = rsqrtf(tot.y * (1.0f / DIMS) - mean * mean + EPSLN);
  float* orow = out + tok * DIMS;
  unsigned short* brow = x1bf + tok * DIMS;
#pragma unroll
  for (int i = 0; i < 2; ++i) {
    int d = (tid << 1) + (i << 9);
    float v0 = (row[d]     - mean) * inv * g0[d]     + b0[d];
    float v1 = (row[d + 1] - mean) * inv * g0[d + 1] + b0[d + 1];
    orow[d] = v0; orow[d + 1] = v1;
    *(unsigned*)(brow + d) = (unsigned)f2bf(v0) | ((unsigned)f2bf(v1) << 16);
  }
}

// ---------------- Final: LN(x1 + ff) in place on d_out ----------------------
__global__ __launch_bounds__(256) void holo_final_ln(
    const float* __restrict__ ff,
    const float* __restrict__ g1, const float* __restrict__ b1,
    float* __restrict__ io)
{
  __shared__ float2 red[256];
  const int tid = threadIdx.x;
  const size_t tok = blockIdx.x;
  const float* fr = ff + tok * DIMS;
  float* xr = io + tok * DIMS;
  float vals[4]; float s = 0.f, ss = 0.f;
#pragma unroll
  for (int i = 0; i < 4; ++i) {
    int d = tid + (i << 8);
    float v = xr[d] + fr[d];
    vals[i] = v; s += v; ss += v * v;
  }
  float2 tot = block_reduce2(make_float2(s, ss), red);
  float mean = tot.x * (1.0f / DIMS);
  float inv  = rsqrtf(tot.y * (1.0f / DIMS) - mean * mean + EPSLN);
#pragma unroll
  for (int i = 0; i < 4; ++i) {
    int d = tid + (i << 8);
    xr[d] = (vals[i] - mean) * inv * g1[d] + b1[d];
  }
}

// ---------------- launch ----------------------------------------------------
extern "C" void kernel_launch(void* const* d_in, const int* in_sizes, int n_in,
                              void* d_out, int out_size, void* d_ws, size_t ws_size,
                              hipStream_t stream) {
  (void)in_sizes; (void)n_in; (void)out_size; (void)ws_size;
  const float* x     = (const float*)d_in[0];
  const float* Wq    = (const float*)d_in[1];
  const float* bq    = (const float*)d_in[2];
  const float* gq    = (const float*)d_in[3];
  const float* betaq = (const float*)d_in[4];
  const float* Wk    = (const float*)d_in[5];
  const float* bk    = (const float*)d_in[6];
  const float* gk    = (const float*)d_in[7];
  const float* betak = (const float*)d_in[8];
  const float* g0    = (const float*)d_in[9];
  const float* beta0 = (const float*)d_in[10];
  const float* W1    = (const float*)d_in[11];
  const float* b1    = (const float*)d_in[12];
  const float* W2    = (const float*)d_in[13];
  const float* b2    = (const float*)d_in[14];
  const float* g1    = (const float*)d_in[15];
  const float* beta1 = (const float*)d_in[16];
  float* out = (float*)d_out;
  float* ws  = (float*)d_ws;

  const size_t MT = (size_t)MTOK;
  // f32 scratch
  float* qkbuf = ws;                          // MT*QKSTR      [dead after stage A]
  float* xbbuf = qkbuf + MT * QKSTR;          // MT*NFPAD      [dead after stage B]
  float* qfbuf = xbbuf + MT * NFPAD;          // MT*NFPAD      [dead after stage B]
  float* sfbuf = qfbuf + MT * NFPAD;          // BATCH*NFPAD
  float* bqk   = sfbuf + BATCH * NFPAD;       // QKSTR
  // aliases over dead regions
  unsigned short* hbf = (unsigned short*)ws;  // MT*FFD bf16 == qk region exactly
  float* ffbuf = xbbuf;                       // MT*DIMS f32 fits in XB slot
  // bf16 staging after the f32 region
  float* tail = bqk + QKSTR + 1024;
  unsigned short* xbf  = (unsigned short*)tail;                 // MT*DIMS
  unsigned short* x1bf = xbf  + MT * DIMS;                      // MT*DIMS
  unsigned short* wqkb = x1bf + MT * DIMS;                      // QKSTR*DIMS
  unsigned short* w1b  = wqkb + (size_t)QKSTR * DIMS;           // FFD*DIMS
  unsigned short* w2b  = w1b  + (size_t)FFD * DIMS;             // DIMS*FFD

  // one-time bf16 staging + fused q|k weight/bias concat
  cvt_bf16<<<(MTOK * DIMS) / 2048, 256, 0, stream>>>(x,  xbf, MTOK * DIMS);
  cvt_bf16<<<(DIMS * DIMS) / 2048, 256, 0, stream>>>(Wq, wqkb, DIMS * DIMS);
  cvt_bf16<<<(DIMS * DIMS) / 2048, 256, 0, stream>>>(Wk, wqkb + (size_t)DIMS * DIMS, DIMS * DIMS);
  cvt_bf16<<<(FFD * DIMS) / 2048, 256, 0, stream>>>(W1, w1b, FFD * DIMS);
  cvt_bf16<<<(FFD * DIMS) / 2048, 256, 0, stream>>>(W2, w2b, FFD * DIMS);
  copy_f32<<<DIMS / 256, 256, 0, stream>>>(bq, bqk, DIMS);
  copy_f32<<<DIMS / 256, 256, 0, stream>>>(bk, bqk + DIMS, DIMS);

  const size_t shmem = 2 * BUF_BYTES;   // 51200 B, double-buffered
  // fused q|k projection: one pass over xbf, N = 2048
  gemm_bf16<<<dim3(MTOK / GTM, QKSTR / GTN), 256, shmem, stream>>>(xbf, wqkb, bqk, qkbuf, MTOK, QKSTR, DIMS, 0, 0);
  // holographic mixer in frequency domain
  holo_stage_a<<<MTOK, 256, 0, stream>>>(x, qkbuf, gq, betaq, gk, betak, xbbuf, qfbuf);
  holo_sum_xb<<<dim3((2 * NFREQ + 255) / 256, BATCH), 256, 0, stream>>>(xbbuf, sfbuf);
  holo_stage_b<<<MTOK, 256, 0, stream>>>(x, xbbuf, qfbuf, sfbuf, g0, beta0, out, x1bf);
  // feed-forward (h kept in bf16, written directly by FFN1 epilogue)
  gemm_bf16<<<dim3(MTOK / GTM, FFD / GTN), 256, shmem, stream>>>(x1bf, w1b, b1, hbf, MTOK, FFD, DIMS, 1, 1);
  gemm_bf16<<<dim3(MTOK / GTM, DIMS / GTN), 256, shmem, stream>>>(hbf, w2b, b2, ffbuf, MTOK, DIMS, FFD, 0, 0);
  holo_final_ln<<<MTOK, 256, 0, stream>>>(ffbuf, g1, beta1, out);
}
